// CCASSGEncoder_70007966924832
// MI455X (gfx1250) — compile-verified, matrix operand order checked by reading, not run on hardware
//
#include <hip/hip_runtime.h>
#include <hip/hip_bf16.h>
#include <stdint.h>

#define D 128
#define TILE_M 16

// Device-side bf16 vector types for WMMA (host only ever sees __hip_bfloat16*).
typedef __attribute__((ext_vector_type(16))) __bf16 v16bf;
typedef __attribute__((ext_vector_type(8)))  __bf16 v8bf;
typedef __attribute__((ext_vector_type(8)))  float  v8f;

// ---------------- elementwise / graph helpers ----------------

__global__ void k_zero_f32(float* __restrict__ p, int n) {
  int i = blockIdx.x * blockDim.x + threadIdx.x;
  if (i < n) p[i] = 0.0f;
}

__global__ void k_degrees(const int* __restrict__ src, const int* __restrict__ dst,
                          float* __restrict__ outdeg, float* __restrict__ indeg, int E) {
  int e = blockIdx.x * blockDim.x + threadIdx.x;
  if (e < E) {
    atomicAdd(&outdeg[src[e]], 1.0f);
    atomicAdd(&indeg [dst[e]], 1.0f);
  }
}

__global__ void k_rsqrt_inplace(float* __restrict__ p, int n) {
  int i = blockIdx.x * blockDim.x + threadIdx.x;
  if (i < n) p[i] = rsqrtf(fmaxf(p[i], 1.0f));
}

__global__ void k_cvt_bf16(const float* __restrict__ w, __hip_bfloat16* __restrict__ wb, int n) {
  int i = blockIdx.x * blockDim.x + threadIdx.x;
  if (i < n) {
    __bf16 h = (__bf16)w[i];
    reinterpret_cast<__bf16*>(wb)[i] = h;
  }
}

// xb = bf16(x * norm_src[row])
__global__ void k_scale_cvt(const float* __restrict__ x, const float* __restrict__ ns,
                            __hip_bfloat16* __restrict__ xb, int n) {
  int i = blockIdx.x * blockDim.x + threadIdx.x;
  if (i < n) {
    __bf16 h = (__bf16)(x[i] * ns[i >> 7]);
    reinterpret_cast<__bf16*>(xb)[i] = h;
  }
}

// Fused layer-1 finalize + layer-2 input prep:
// xb = bf16( relu(agg*norm_dst[r] + b[j]) * norm_src[r] )
__global__ void k_relu_rescale_cvt(const float* __restrict__ agg, const float* __restrict__ ndv,
                                   const float* __restrict__ ns, const float* __restrict__ b,
                                   __hip_bfloat16* __restrict__ xb, int n) {
  int i = blockIdx.x * blockDim.x + threadIdx.x;
  if (i < n) {
    int r = i >> 7, j = i & (D - 1);
    float v = fmaxf(fmaf(agg[i], ndv[r], b[j]), 0.0f);
    __bf16 h = (__bf16)(v * ns[r]);
    reinterpret_cast<__bf16*>(xb)[i] = h;
  }
}

__global__ void k_finalize(const float* __restrict__ agg, const float* __restrict__ ndv,
                           const float* __restrict__ b, float* __restrict__ out, int n) {
  int i = blockIdx.x * blockDim.x + threadIdx.x;
  if (i < n) {
    int r = i >> 7, j = i & (D - 1);
    out[i] = fmaxf(fmaf(agg[i], ndv[r], b[j]), 0.0f);
  }
}

// agg[dst[e]][:] += H[src[e]][:], one wave per edge: each lane owns 4
// consecutive floats (b128 load + 4 f32 atomics). src/dst are wave-uniform.
// H/agg are 51.2 MB each -> resident in the 192 MB L2, atomics run at L2 rate.
__global__ void k_scatter(const float* __restrict__ H, const int* __restrict__ src,
                          const int* __restrict__ dst, float* __restrict__ agg, int E) {
  int wid = (blockIdx.x * blockDim.x + threadIdx.x) >> 5;  // edge index
  if (wid >= E) return;
  const int lane = threadIdx.x & 31;
  const int s = src[wid];
  const int d = dst[wid];
  const float4 v = *(const float4*)(H + (size_t)s * D + lane * 4);
  float* ap = agg + (size_t)d * D + lane * 4;
  atomicAdd(ap + 0, v.x);
  atomicAdd(ap + 1, v.y);
  atomicAdd(ap + 2, v.z);
  atomicAdd(ap + 3, v.w);
}

// ---------------- WMMA GEMM: H[N,128] = Xb[N,128] @ Wb[128,128] ----------------
// 256 threads = 8 waves; wave w owns output columns [w*16, w*16+16).
// B fragments (all 4 K-chunks of the 128x128 weight) are loaded once per block
// and reused while the block grid-strides over 16-row M tiles. All 8 b128
// A-loads for a tile are hoisted ahead of the 4 chained WMMAs so they issue as
// one load clause with a single wait.
__global__ void k_gemm_bf16_wmma(const __hip_bfloat16* __restrict__ Ah,
                                 const __hip_bfloat16* __restrict__ Wh,
                                 float* __restrict__ H, int ntiles) {
  const __bf16* __restrict__ A  = reinterpret_cast<const __bf16*>(Ah);
  const __bf16* __restrict__ Wb = reinterpret_cast<const __bf16*>(Wh);

  const int lane  = threadIdx.x & 31;
  const int wave  = threadIdx.x >> 5;     // 0..7 -> n0 = wave*16
  const int lrow  = lane & 15;
  const int lhalf = lane >> 4;            // 0: K 0-7/16-23 ; 1: K 8-15/24-31
  const int c     = wave * 16 + lrow;     // output column owned by this lane

  // B fragment per K-chunk: lane holds column c, K per 16-bit A/B layout.
  v16bf bf[4];
#pragma unroll
  for (int kc = 0; kc < 4; ++kc) {
    const int kbase = kc * 32 + lhalf * 8;
#pragma unroll
    for (int i = 0; i < 8; ++i) {
      bf[kc][i]     = Wb[(kbase + i)      * D + c];
      bf[kc][i + 8] = Wb[(kbase + 16 + i) * D + c];
    }
  }

  for (int t = blockIdx.x; t < ntiles; t += gridDim.x) {
    const __bf16* abase = A + (size_t)(t * TILE_M + lrow) * D + lhalf * 8;

    if (t + gridDim.x < ntiles) {
      __builtin_prefetch(A + (size_t)((t + gridDim.x) * TILE_M + lrow) * D, 0, 1);
    }

    // Hoist all A-fragment loads: one clause of 8x global_load_b128.
    v8bf areg[8];
#pragma unroll
    for (int kc = 0; kc < 4; ++kc) {
      areg[2 * kc]     = *(const v8bf*)(abase + kc * 32);       // K kbase..kbase+7
      areg[2 * kc + 1] = *(const v8bf*)(abase + kc * 32 + 16);  // K kbase+16..+23
    }

    v8f acc = {};
#pragma unroll
    for (int kc = 0; kc < 4; ++kc) {
      v16bf a;
#pragma unroll
      for (int i = 0; i < 8; ++i) {
        a[i]     = areg[2 * kc][i];
        a[i + 8] = areg[2 * kc + 1][i];
      }
      acc = __builtin_amdgcn_wmma_f32_16x16x32_bf16(
          /*neg_a=*/false, a, /*neg_b=*/false, bf[kc],
          /*c_mod=*/(short)0, acc, /*reuse_a=*/false, /*reuse_b=*/false);
    }

    // D layout: VGPR v -> row (v + 8*lhalf), col = c
    float* hbase = H + (size_t)(t * TILE_M + lhalf * 8) * D + c;
#pragma unroll
    for (int v = 0; v < 8; ++v) hbase[v * D] = acc[v];
  }
}

// ---------------- launch ----------------

extern "C" void kernel_launch(void* const* d_in, const int* in_sizes, int n_in,
                              void* d_out, int out_size, void* d_ws, size_t ws_size,
                              hipStream_t stream) {
  const float* x  = (const float*)d_in[0];
  const float* W1 = (const float*)d_in[1];
  const float* b1 = (const float*)d_in[2];
  const float* W2 = (const float*)d_in[3];
  const float* b2 = (const float*)d_in[4];
  const int*  src = (const int*)d_in[5];
  const int*  dst = (const int*)d_in[6];

  const int N  = in_sizes[0] / D;   // 100000
  const int E  = in_sizes[5];       // 1.6M
  const int ND = N * D;             // 12.8M

  // Workspace carve (~129 MB): norms, bf16 weights, bf16 activations, H, agg.
  char* w = (char*)d_ws;
  auto carve = [&](size_t bytes) -> void* {
    void* p = (void*)w;
    w += (bytes + 255) & ~(size_t)255;
    return p;
  };
  float*          norm_src = (float*)carve((size_t)N * 4);
  float*          norm_dst = (float*)carve((size_t)N * 4);
  __hip_bfloat16* Wb1      = (__hip_bfloat16*)carve((size_t)D * D * 2);
  __hip_bfloat16* Wb2      = (__hip_bfloat16*)carve((size_t)D * D * 2);
  __hip_bfloat16* Xb       = (__hip_bfloat16*)carve((size_t)ND * 2);
  float*          H        = (float*)carve((size_t)ND * 4);
  float*          agg      = (float*)carve((size_t)ND * 4);

  const int T = 256;
  auto cdiv = [](int a, int b) { return (a + b - 1) / b; };

  // Degrees -> rsqrt norms (shared by both layers).
  k_zero_f32<<<cdiv(N, T), T, 0, stream>>>(norm_src, N);
  k_zero_f32<<<cdiv(N, T), T, 0, stream>>>(norm_dst, N);
  k_degrees<<<cdiv(E, T), T, 0, stream>>>(src, dst, norm_src, norm_dst, E);
  k_rsqrt_inplace<<<cdiv(N, T), T, 0, stream>>>(norm_src, N);
  k_rsqrt_inplace<<<cdiv(N, T), T, 0, stream>>>(norm_dst, N);

  // Weights to bf16.
  k_cvt_bf16<<<cdiv(D * D, T), T, 0, stream>>>(W1, Wb1, D * D);
  k_cvt_bf16<<<cdiv(D * D, T), T, 0, stream>>>(W2, Wb2, D * D);

  const int ntiles = N / TILE_M;                      // N is a multiple of 16
  const int GB     = ntiles < 1536 ? ntiles : 1536;   // grid-stride GEMM blocks
  const int scatterBlocks = cdiv(E * 32, T);          // one wave per edge

  // ---- layer 1 ----
  k_scale_cvt<<<cdiv(ND, T), T, 0, stream>>>(x, norm_src, Xb, ND);
  k_gemm_bf16_wmma<<<GB, T, 0, stream>>>(Xb, Wb1, H, ntiles);
  k_zero_f32<<<cdiv(ND, T), T, 0, stream>>>(agg, ND);
  k_scatter<<<scatterBlocks, T, 0, stream>>>(H, src, dst, agg, E);
  k_relu_rescale_cvt<<<cdiv(ND, T), T, 0, stream>>>(agg, norm_dst, norm_src, b1, Xb, ND);

  // ---- layer 2 ----
  k_gemm_bf16_wmma<<<GB, T, 0, stream>>>(Xb, Wb2, H, ntiles);
  k_zero_f32<<<cdiv(ND, T), T, 0, stream>>>(agg, ND);
  k_scatter<<<scatterBlocks, T, 0, stream>>>(H, src, dst, agg, E);
  k_finalize<<<cdiv(ND, T), T, 0, stream>>>(agg, norm_dst, b2, (float*)d_out, ND);
}
